// drGAT_51823075394018
// MI455X (gfx1250) — matrix operand order repaired and checked by reading, not compiled, not run on hardware
//
#include <hip/hip_runtime.h>
#include <hip/hip_bf16.h>
#include <math.h>

// ---------------- problem constants ----------------
#define NDRUG 500
#define NCELL 500
#define NGENE 9000
#define NN    10000          // total nodes
#define EE    200000         // edges
#define E2    210000         // edges + self loops
#define BATCH 8192
#define H1    256
#define HEADS 4
#define CH    128
#define D1    512            // HEADS*CH
#define LRELU 0.2f
#define LN_EPS 1e-5f

typedef __attribute__((ext_vector_type(16))) __bf16 v16bf;
typedef __attribute__((ext_vector_type(8)))  float  v8f;

union FragBF { v16bf v; unsigned short h[16]; unsigned u[8]; uint4 q[2]; };

__device__ __forceinline__ unsigned short f2bf(float f) {
    union { float f; unsigned u; } c; c.f = f;
    return (unsigned short)(c.u >> 16);   // truncate fp32 -> bf16
}
__device__ __forceinline__ unsigned pkbf(float a, float b) {
    return (unsigned)f2bf(a) | ((unsigned)f2bf(b) << 16);
}

__device__ __forceinline__ float waveReduceSum(float v) {
    #pragma unroll
    for (int off = 16; off > 0; off >>= 1) v += __shfl_down(v, off);
    return v;
}

// monotonic float<->uint encoding for atomicMax on floats
__device__ __forceinline__ unsigned fenc(float f) {
    union { float f; unsigned u; } c; c.f = f;
    return (c.u & 0x80000000u) ? ~c.u : (c.u | 0x80000000u);
}
__device__ __forceinline__ float fdec(unsigned u) {
    union { float f; unsigned u; } c;
    c.u = (u & 0x80000000u) ? (u ^ 0x80000000u) : ~u;
    return c.f;
}

// ---------------- WMMA GEMM: C[M,Nout] = A[M,K] * W[Nout,K]^T (+bias) ----------------
// block = 256 threads = 8 waves. Wave w computes a 16-row x 64-col tile:
// rows (blockIdx.x*8+w)*16, cols blockIdx.y*64 (four 16x16 WMMA tiles sharing the A frag).
// B tile (32K x 64N) staged in LDS as packed bf16; main K loop is guard-free b128 loads.
__global__ __launch_bounds__(256)
void gemm_wmma_bf16(const float* __restrict__ A, int lda,
                    const float* __restrict__ W,       // [Nout, K] row-major
                    const float* __restrict__ bias,    // may be null
                    float* __restrict__ C, int ldc,
                    int M, int K, int Nout)
{
    __shared__ unsigned Bs[64][20];   // packed bf16 pairs; 16 used + pad (80B row stride)
    const int lane = threadIdx.x & 31;
    const int wave = threadIdx.x >> 5;
    const int row0 = (blockIdx.x * 8 + wave) * 16;
    const int n0   = blockIdx.y * 64;
    const int rloc = lane & 15;
    const int hA   = (lane & 16) ? 16 : 0;  // A: hi half-wave holds K=16..31
    const int hB   = (lane & 16) ? 8 : 0;   // B: hi half-wave holds K=16..31 (uint units)

    int ra = row0 + rloc; if (ra >= M) ra = M - 1;
    const float* __restrict__ ar = A + (size_t)ra * lda;

    // B staging: thread t -> row sn (0..63), k offset sk (0,8,16,24); 8 floats each
    const int sn = threadIdx.x >> 2;
    const int sk = (threadIdx.x & 3) << 3;
    const float* __restrict__ wrow = W + (size_t)(n0 + sn) * K;

    v8f acc0 = {}, acc1 = {}, acc2 = {}, acc3 = {};
    const int K32 = K & ~31;

    for (int k0 = 0; k0 < K32; k0 += 32) {
        // ---- stage B tile: 2x b128 global loads -> 1x ds_store_b128 (packed bf16) ----
        float4 w0 = *(const float4*)(wrow + k0 + sk);
        float4 w1 = *(const float4*)(wrow + k0 + sk + 4);
        uint4 wp;
        wp.x = pkbf(w0.x, w0.y); wp.y = pkbf(w0.z, w0.w);
        wp.z = pkbf(w1.x, w1.y); wp.w = pkbf(w1.z, w1.w);
        *(uint4*)&Bs[sn][sk >> 1] = wp;
        __syncthreads();

        if (k0 + 32 < K) __builtin_prefetch(ar + k0 + 32, 0, 1);  // global_prefetch_b8

        // ---- A fragment: one contiguous 16-float run per lane -> 4x b128 loads ----
        FragBF af;
        float4 a0 = *(const float4*)(ar + k0 + hA);
        float4 a1 = *(const float4*)(ar + k0 + hA + 4);
        float4 a2 = *(const float4*)(ar + k0 + hA + 8);
        float4 a3 = *(const float4*)(ar + k0 + hA + 12);
        af.u[0] = pkbf(a0.x, a0.y); af.u[1] = pkbf(a0.z, a0.w);
        af.u[2] = pkbf(a1.x, a1.y); af.u[3] = pkbf(a1.z, a1.w);
        af.u[4] = pkbf(a2.x, a2.y); af.u[5] = pkbf(a2.z, a2.w);
        af.u[6] = pkbf(a3.x, a3.y); af.u[7] = pkbf(a3.z, a3.w);

        // ---- B fragments: contiguous packed pairs -> 2x ds_load_b128 each ----
        FragBF b0, b1, b2, b3;
        b0.q[0] = *(const uint4*)&Bs[rloc     ][hB];
        b0.q[1] = *(const uint4*)&Bs[rloc     ][hB + 4];
        b1.q[0] = *(const uint4*)&Bs[rloc + 16][hB];
        b1.q[1] = *(const uint4*)&Bs[rloc + 16][hB + 4];
        b2.q[0] = *(const uint4*)&Bs[rloc + 32][hB];
        b2.q[1] = *(const uint4*)&Bs[rloc + 32][hB + 4];
        b3.q[0] = *(const uint4*)&Bs[rloc + 48][hB];
        b3.q[1] = *(const uint4*)&Bs[rloc + 48][hB + 4];

        acc0 = __builtin_amdgcn_wmma_f32_16x16x32_bf16(
                   false, af.v, false, b0.v, (short)0, acc0, false, false);
        acc1 = __builtin_amdgcn_wmma_f32_16x16x32_bf16(
                   false, af.v, false, b1.v, (short)0, acc1, false, false);
        acc2 = __builtin_amdgcn_wmma_f32_16x16x32_bf16(
                   false, af.v, false, b2.v, (short)0, acc2, false, false);
        acc3 = __builtin_amdgcn_wmma_f32_16x16x32_bf16(
                   false, af.v, false, b3.v, (short)0, acc3, false, false);
        __syncthreads();
    }

    // ---- K tail (one iteration, clamped-index loads, no exec-divergent branches) ----
    if (K32 < K) {
        const int k0 = K32;
        float wv[8];
        #pragma unroll
        for (int i = 0; i < 8; ++i) {
            int gk = k0 + sk + i;
            int gc = gk < K ? gk : K - 1;
            wv[i] = wrow[gc] * (gk < K ? 1.f : 0.f);
        }
        uint4 wp;
        wp.x = pkbf(wv[0], wv[1]); wp.y = pkbf(wv[2], wv[3]);
        wp.z = pkbf(wv[4], wv[5]); wp.w = pkbf(wv[6], wv[7]);
        *(uint4*)&Bs[sn][sk >> 1] = wp;
        __syncthreads();

        FragBF af, b0, b1, b2, b3;
        #pragma unroll
        for (int v = 0; v < 8; ++v) {
            int kk = hA + (v << 1);
            int g0 = k0 + kk, g1 = g0 + 1;
            float x = ar[g0 < K ? g0 : K - 1] * (g0 < K ? 1.f : 0.f);
            float y = ar[g1 < K ? g1 : K - 1] * (g1 < K ? 1.f : 0.f);
            af.u[v] = pkbf(x, y);
        }
        b0.q[0] = *(const uint4*)&Bs[rloc     ][hB];
        b0.q[1] = *(const uint4*)&Bs[rloc     ][hB + 4];
        b1.q[0] = *(const uint4*)&Bs[rloc + 16][hB];
        b1.q[1] = *(const uint4*)&Bs[rloc + 16][hB + 4];
        b2.q[0] = *(const uint4*)&Bs[rloc + 32][hB];
        b2.q[1] = *(const uint4*)&Bs[rloc + 32][hB + 4];
        b3.q[0] = *(const uint4*)&Bs[rloc + 48][hB];
        b3.q[1] = *(const uint4*)&Bs[rloc + 48][hB + 4];

        acc0 = __builtin_amdgcn_wmma_f32_16x16x32_bf16(
                   false, af.v, false, b0.v, (short)0, acc0, false, false);
        acc1 = __builtin_amdgcn_wmma_f32_16x16x32_bf16(
                   false, af.v, false, b1.v, (short)0, acc1, false, false);
        acc2 = __builtin_amdgcn_wmma_f32_16x16x32_bf16(
                   false, af.v, false, b2.v, (short)0, acc2, false, false);
        acc3 = __builtin_amdgcn_wmma_f32_16x16x32_bf16(
                   false, af.v, false, b3.v, (short)0, acc3, false, false);
    }

    // ---- store four 16x16 tiles: lane holds col (lane&15); rows v + (hi?8:0) ----
    const int rbase = row0 + ((lane & 16) ? 8 : 0);
    const int na = n0 + rloc;
    const float ba0 = bias ? bias[na]      : 0.f;
    const float ba1 = bias ? bias[na + 16] : 0.f;
    const float ba2 = bias ? bias[na + 32] : 0.f;
    const float ba3 = bias ? bias[na + 48] : 0.f;
    #pragma unroll
    for (int v = 0; v < 8; ++v) {
        int r = rbase + v;
        if (r < M) {
            float* cr = C + (size_t)r * ldc + na;
            cr[0]  = acc0[v] + ba0;
            cr[16] = acc1[v] + ba1;
            cr[32] = acc2[v] + ba2;
            cr[48] = acc3[v] + ba3;
        }
    }
}

// ---------------- fills ----------------
__global__ void fill_f32(float* p, size_t n, float val) {
    size_t i = (size_t)blockIdx.x * blockDim.x + threadIdx.x;
    size_t st = (size_t)gridDim.x * blockDim.x;
    for (; i < n; i += st) p[i] = val;
}
__global__ void fill_u32(unsigned* p, size_t n, unsigned val) {
    size_t i = (size_t)blockIdx.x * blockDim.x + threadIdx.x;
    size_t st = (size_t)gridDim.x * blockDim.x;
    for (; i < n; i += st) p[i] = val;
}

// ---------------- self-loop attr (fill_value='mean') ----------------
__global__ void edge_deg_k(const int* __restrict__ ei, const float* __restrict__ ea,
                           float* cnt, float* asum) {
    int e = blockIdx.x * blockDim.x + threadIdx.x;
    if (e >= EE) return;
    int d = ei[EE + e];
    atomicAdd(&cnt[d], 1.f);
    atomicAdd(&asum[d], ea[e]);
}
__global__ void loop_attr_k(const float* cnt, const float* asum, float* la) {
    int n = blockIdx.x * blockDim.x + threadIdx.x;
    if (n < NN) la[n] = asum[n] / fmaxf(cnt[n], 1.f);
}

// ---------------- attention logits ----------------
__global__ void ledot_k(const float* __restrict__ ledge, const float* __restrict__ aedge,
                        float* ledot) {
    int h = threadIdx.x >> 5, lane = threadIdx.x & 31;
    float s = 0.f;
    for (int c = lane; c < CH; c += 32) s += ledge[h * CH + c] * aedge[h * CH + c];
    s = waveReduceSum(s);
    if (lane == 0) ledot[h] = s;
}

__global__ __launch_bounds__(128)
void attn_al_k(const float* __restrict__ xh, const float* __restrict__ asrc,
               const float* __restrict__ adst, float* alsrc, float* aldst) {
    int n = blockIdx.x;
    int h = threadIdx.x >> 5, lane = threadIdx.x & 31;
    const float* xr = xh + (size_t)n * D1 + h * CH;
    float s1 = 0.f, s2 = 0.f;
    for (int c = lane; c < CH; c += 32) {
        float x = xr[c];
        s1 += x * asrc[h * CH + c];
        s2 += x * adst[h * CH + c];
    }
    s1 = waveReduceSum(s1); s2 = waveReduceSum(s2);
    if (lane == 0) { alsrc[n * HEADS + h] = s1; aldst[n * HEADS + h] = s2; }
}

// ---------------- segment softmax over destination nodes ----------------
__global__ void edge_raw_k(const int* __restrict__ ei, const float* __restrict__ ea,
                           const float* __restrict__ la,
                           const float* __restrict__ alsrc, const float* __restrict__ aldst,
                           const float* __restrict__ ledot,
                           float* alpha, unsigned* m) {
    int t = blockIdx.x * blockDim.x + threadIdx.x;
    if (t >= E2 * HEADS) return;
    int e = t >> 2, h = t & 3;
    int s, d; float eav;
    if (e < EE) { s = ei[e]; d = ei[EE + e]; eav = ea[e]; }
    else        { s = d = e - EE;            eav = la[s]; }
    float r = alsrc[s * HEADS + h] + aldst[d * HEADS + h] + eav * ledot[h];
    r = (r > 0.f) ? r : LRELU * r;
    alpha[t] = r;
    atomicMax(&m[d * HEADS + h], fenc(r));
}
__global__ void edge_exp_k(const int* __restrict__ ei, const unsigned* __restrict__ m,
                           float* alpha, float* ssum) {
    int t = blockIdx.x * blockDim.x + threadIdx.x;
    if (t >= E2 * HEADS) return;
    int e = t >> 2, h = t & 3;
    int d = (e < EE) ? ei[EE + e] : (e - EE);
    float ex = __expf(alpha[t] - fdec(m[d * HEADS + h]));
    alpha[t] = ex;
    atomicAdd(&ssum[d * HEADS + h], ex);
}
__global__ void edge_norm_k(const int* __restrict__ ei, const float* __restrict__ ssum,
                            float* alpha) {
    int t = blockIdx.x * blockDim.x + threadIdx.x;
    if (t >= E2 * HEADS) return;
    int e = t >> 2, h = t & 3;
    int d = (e < EE) ? ei[EE + e] : (e - EE);
    alpha[t] = alpha[t] / (ssum[d * HEADS + h] + 1e-16f);
}

// ---------------- aggregation: agg[d] += xh[s] * alpha ----------------
__global__ void init_bias_k(float* agg, const float* __restrict__ bias) {
    size_t i = (size_t)blockIdx.x * blockDim.x + threadIdx.x;
    if (i < (size_t)NN * D1) agg[i] = bias[i & (D1 - 1)];
}
__global__ __launch_bounds__(128)
void edge_agg_k(const int* __restrict__ ei, const float* __restrict__ xh,
                const float* __restrict__ alpha, float* agg) {
    int e = blockIdx.x;
    int s, d;
    if (e < EE) { s = ei[e]; d = ei[EE + e]; } else { s = d = e - EE; }
    int j0 = threadIdx.x * 4;                 // 128 threads x 4 = 512 channels
    int h  = j0 >> 7;
    float a = alpha[(size_t)e * HEADS + h];
    float4 xv = *(const float4*)(xh + (size_t)s * D1 + j0);
    float* ag = agg + (size_t)d * D1 + j0;
    atomicAdd(&ag[0], xv.x * a);
    atomicAdd(&ag[1], xv.y * a);
    atomicAdd(&ag[2], xv.z * a);
    atomicAdd(&ag[3], xv.w * a);
}

// ---------------- LayerNorm + ReLU, in place, one node per block ----------------
__global__ __launch_bounds__(128)
void ln_relu_k(float* x, const float* __restrict__ g, const float* __restrict__ b) {
    __shared__ float red[128];
    int n = blockIdx.x, t = threadIdx.x;
    float* xr = x + (size_t)n * D1;
    float v4[4]; float s = 0.f;
    #pragma unroll
    for (int i = 0; i < 4; ++i) { v4[i] = xr[t * 4 + i]; s += v4[i]; }
    red[t] = s; __syncthreads();
    for (int off = 64; off > 0; off >>= 1) { if (t < off) red[t] += red[t + off]; __syncthreads(); }
    float mu = red[0] / (float)D1;
    __syncthreads();
    float vs = 0.f;
    #pragma unroll
    for (int i = 0; i < 4; ++i) { float dd = v4[i] - mu; vs += dd * dd; }
    red[t] = vs; __syncthreads();
    for (int off = 64; off > 0; off >>= 1) { if (t < off) red[t] += red[t + off]; __syncthreads(); }
    float inv = rsqrtf(red[0] / (float)D1 + LN_EPS);
    #pragma unroll
    for (int i = 0; i < 4; ++i) {
        int j = t * 4 + i;
        float y = (v4[i] - mu) * inv * g[j] + b[j];
        xr[j] = fmaxf(y, 0.f);
    }
}

// ---------------- dense attention matrix scatter ----------------
__global__ void att_write_k(const int* __restrict__ ei, const float* __restrict__ a0,
                            const float* __restrict__ a1, float* att) {
    int e = blockIdx.x * blockDim.x + threadIdx.x;
    if (e >= E2) return;
    int s, d;
    if (e < EE) { s = ei[e]; d = ei[EE + e]; } else { s = d = e - EE; }
    size_t i4 = (size_t)e * 4;
    float m0 = 0.25f * (a0[i4] + a0[i4 + 1] + a0[i4 + 2] + a0[i4 + 3]);
    float m1 = 0.25f * (a1[i4] + a1[i4 + 1] + a1[i4 + 2] + a1[i4 + 3]);
    att[(size_t)s * NN + d] = m0 + m1;
}

// ---------------- final pair MLP: one wave per batch element ----------------
__global__ __launch_bounds__(256)
void mlp_k(const float* __restrict__ xf, const int* __restrict__ idxd,
           const int* __restrict__ idxc, const float* __restrict__ w,
           const float* __restrict__ bp, float* pred) {
    int wv = threadIdx.x >> 5, lane = threadIdx.x & 31;
    int bi = blockIdx.x * 8 + wv;
    if (bi >= BATCH) return;
    const float* xd = xf + (size_t)idxd[bi] * D1;
    const float* xc = xf + (size_t)idxc[bi] * D1;
    float s = 0.f;
    for (int j = lane; j < D1; j += 32) s += xd[j] * w[j] + xc[j] * w[D1 + j];
    s = waveReduceSum(s);
    if (lane == 0) pred[bi] = s + bp[0];
}

// ---------------- host side ----------------
static inline void launch_gemm(const float* A, int lda, const float* W,
                               const float* bias, float* C, int ldc,
                               int M, int K, int Nout, hipStream_t s) {
    int tm = (M + 15) / 16;
    dim3 g((tm + 7) / 8, Nout / 64);
    gemm_wmma_bf16<<<g, 256, 0, s>>>(A, lda, W, bias, C, ldc, M, K, Nout);
}

extern "C" void kernel_launch(void* const* d_in, const int* in_sizes, int n_in,
                              void* d_out, int out_size, void* d_ws, size_t ws_size,
                              hipStream_t stream) {
    (void)in_sizes; (void)n_in; (void)out_size; (void)ws_size;
    const float* drug   = (const float*)d_in[0];
    const float* cell   = (const float*)d_in[1];
    const float* gene   = (const float*)d_in[2];
    const int*   ei     = (const int*)  d_in[3];
    const float* ea     = (const float*)d_in[4];
    const int*   idxd   = (const int*)  d_in[5];
    const int*   idxc   = (const int*)  d_in[6];
    const float* Wd = (const float*)d_in[7];  const float* bd = (const float*)d_in[8];
    const float* Wc = (const float*)d_in[9];  const float* bc = (const float*)d_in[10];
    const float* Wg = (const float*)d_in[11]; const float* bg = (const float*)d_in[12];
    const float* l0_lin  = (const float*)d_in[13];
    const float* l0_asrc = (const float*)d_in[14];
    const float* l0_adst = (const float*)d_in[15];
    const float* l0_led  = (const float*)d_in[16];
    const float* l0_aed  = (const float*)d_in[17];
    const float* l0_bias = (const float*)d_in[18];
    const float* l0_lng  = (const float*)d_in[19];
    const float* l0_lnb  = (const float*)d_in[20];
    const float* l1_lin  = (const float*)d_in[21];
    const float* l1_asrc = (const float*)d_in[22];
    const float* l1_adst = (const float*)d_in[23];
    const float* l1_led  = (const float*)d_in[24];
    const float* l1_aed  = (const float*)d_in[25];
    const float* l1_bias = (const float*)d_in[26];
    const float* l1_lng  = (const float*)d_in[27];
    const float* l1_lnb  = (const float*)d_in[28];
    const float* mlp_w   = (const float*)d_in[29];
    const float* mlp_b   = (const float*)d_in[30];

    float* out  = (float*)d_out;          // [0..8191] pred, then 10000x10000 att
    float* att  = out + BATCH;

    // workspace layout (floats)
    float* ws     = (float*)d_ws;
    float* x0     = ws;                              // NN*H1  = 2,560,000
    float* xh     = x0     + (size_t)NN * H1;        // NN*D1  = 5,120,000 (layer0 then layer1)
    float* agg0   = xh     + (size_t)NN * D1;        // NN*D1
    float* agg1   = agg0   + (size_t)NN * D1;        // NN*D1
    float* alpha0 = agg1   + (size_t)NN * D1;        // E2*HEADS = 840,000
    float* alpha1 = alpha0 + (size_t)E2 * HEADS;     // 840,000
    float* alsrc  = alpha1 + (size_t)E2 * HEADS;     // NN*HEADS
    float* aldst  = alsrc  + (size_t)NN * HEADS;
    unsigned* mbuf = (unsigned*)(aldst + (size_t)NN * HEADS);
    float* ssum   = (float*)(mbuf + (size_t)NN * HEADS);
    float* cnt    = ssum   + (size_t)NN * HEADS;
    float* asum   = cnt    + NN;
    float* la     = asum   + NN;
    float* ledot  = la     + NN;                     // 4

    const unsigned NEG_INF_ENC = 0x007FFFFFu;        // fenc(-inf)
    const int TPB = 256;

    // zero full output (pred + 400MB attention matrix)
    size_t outN = (size_t)BATCH + (size_t)NN * NN;
    fill_f32<<<8192, TPB, 0, stream>>>(out, outN, 0.f);

    // self-loop attrs
    fill_f32<<<40, TPB, 0, stream>>>(cnt,  NN, 0.f);
    fill_f32<<<40, TPB, 0, stream>>>(asum, NN, 0.f);
    edge_deg_k<<<(EE + TPB - 1) / TPB, TPB, 0, stream>>>(ei, ea, cnt, asum);
    loop_attr_k<<<(NN + TPB - 1) / TPB, TPB, 0, stream>>>(cnt, asum, la);

    // input projections -> x0 [NN, 256]
    launch_gemm(drug, NDRUG, Wd, bd, x0,                     H1, NDRUG, NDRUG, H1, stream);
    launch_gemm(cell, NCELL, Wc, bc, x0 + (size_t)NDRUG*H1,  H1, NCELL, NCELL, H1, stream);
    launch_gemm(gene, NGENE, Wg, bg, x0 + (size_t)1000*H1,   H1, NGENE, NGENE, H1, stream);

    // ---- GAT layer 0 ----
    launch_gemm(x0, H1, l0_lin, nullptr, xh, D1, NN, H1, D1, stream);
    attn_al_k<<<NN, 128, 0, stream>>>(xh, l0_asrc, l0_adst, alsrc, aldst);
    ledot_k<<<1, 128, 0, stream>>>(l0_led, l0_aed, ledot);
    fill_u32<<<40, TPB, 0, stream>>>(mbuf, (size_t)NN * HEADS, NEG_INF_ENC);
    fill_f32<<<40, TPB, 0, stream>>>(ssum, (size_t)NN * HEADS, 0.f);
    int gEH = (E2 * HEADS + TPB - 1) / TPB;
    edge_raw_k <<<gEH, TPB, 0, stream>>>(ei, ea, la, alsrc, aldst, ledot, alpha0, mbuf);
    edge_exp_k <<<gEH, TPB, 0, stream>>>(ei, mbuf, alpha0, ssum);
    edge_norm_k<<<gEH, TPB, 0, stream>>>(ei, ssum, alpha0);
    init_bias_k<<<((size_t)NN * D1 + TPB - 1) / TPB, TPB, 0, stream>>>(agg0, l0_bias);
    edge_agg_k<<<E2, 128, 0, stream>>>(ei, xh, alpha0, agg0);
    ln_relu_k<<<NN, 128, 0, stream>>>(agg0, l0_lng, l0_lnb);

    // ---- GAT layer 1 ----
    launch_gemm(agg0, D1, l1_lin, nullptr, xh, D1, NN, D1, D1, stream);
    attn_al_k<<<NN, 128, 0, stream>>>(xh, l1_asrc, l1_adst, alsrc, aldst);
    ledot_k<<<1, 128, 0, stream>>>(l1_led, l1_aed, ledot);
    fill_u32<<<40, TPB, 0, stream>>>(mbuf, (size_t)NN * HEADS, NEG_INF_ENC);
    fill_f32<<<40, TPB, 0, stream>>>(ssum, (size_t)NN * HEADS, 0.f);
    edge_raw_k <<<gEH, TPB, 0, stream>>>(ei, ea, la, alsrc, aldst, ledot, alpha1, mbuf);
    edge_exp_k <<<gEH, TPB, 0, stream>>>(ei, mbuf, alpha1, ssum);
    edge_norm_k<<<gEH, TPB, 0, stream>>>(ei, ssum, alpha1);
    init_bias_k<<<((size_t)NN * D1 + TPB - 1) / TPB, TPB, 0, stream>>>(agg1, l1_bias);
    edge_agg_k<<<E2, 128, 0, stream>>>(ei, xh, alpha1, agg1);
    ln_relu_k<<<NN, 128, 0, stream>>>(agg1, l1_lng, l1_lnb);

    // attention matrix + prediction
    att_write_k<<<(E2 + TPB - 1) / TPB, TPB, 0, stream>>>(ei, alpha0, alpha1, att);
    mlp_k<<<BATCH / 8, 256, 0, stream>>>(agg1, idxd, idxc, mlp_w, mlp_b, out);
}
